// GNN_20839181320246
// MI455X (gfx1250) — matrix operand order repaired and checked by reading, not compiled, not run on hardware
//
#include <hip/hip_runtime.h>
#include <hip/hip_bf16.h>

// ---------------------------------------------------------------------------
// CDNA5 / gfx1250 implementation of the reference GNN.
// Core compute: v_wmma_f32_16x16x32_bf16 GEMMs, weights staged in LDS,
// fused-gather GEMMs for the big edge-wise layers (A read directly from the
// feature tensors through the graph indices; no staging buffer), batch-norm
// folded into the weights, fused bias/leaky/BN-stats and fused segment_sum
// atomics.  wave32, 8 waves (256 threads) per block.
// ---------------------------------------------------------------------------

typedef __bf16 bf16;
typedef __attribute__((ext_vector_type(8)))  float v8f;
typedef __attribute__((ext_vector_type(16))) bf16  v16bf;
typedef __attribute__((ext_vector_type(8)))  bf16  v8bf;

#define LEAK 0.01f
#define EPSV 1e-5f
#define NBATCH 30

__device__ __forceinline__ void atomic_add_f(float* p, float v) {
    __hip_atomic_fetch_add(p, v, __ATOMIC_RELAXED, __HIP_MEMORY_SCOPE_AGENT);
}

// ---------------------------------------------------------------------------
// Weight swizzle: fp32 KxN row-major -> bf16 fragment order for WMMA B operand.
// Fragment (kc, nt): 32 lanes x 16 bf16 contiguous; lane l holds column
// n = nt*16 + (l&15), elements e -> k = kc*32 + (l>>4)*8 + (e&7) + (e>>3)*16.  K padded to 32.
// ---------------------------------------------------------------------------
__global__ void swizzle_w(const float* __restrict__ W, int K, int N, int KC,
                          bf16* __restrict__ Wf) {
    int total = KC * (N / 16) * 512;
    for (int idx = blockIdx.x * blockDim.x + threadIdx.x; idx < total;
         idx += gridDim.x * blockDim.x) {
        int e    = idx & 15;
        int lane = (idx >> 4) & 31;
        int t    = idx >> 9;                  // kc*NT + nt
        int NT   = N / 16;
        int nt   = t % NT;
        int kc   = t / NT;
        int k    = kc * 32 + (lane >> 4) * 8 + (e & 7) + (e >> 3) * 16;
        int n    = nt * 16 + (lane & 15);
        float v  = (k < K) ? W[(size_t)k * N + n] : 0.f;
        Wf[idx]  = (bf16)v;
    }
}

// ---------------------------------------------------------------------------
// Batch-norm fold: from per-column (sum, sumsq) stats of the hidden layer,
// compute s = g*rsqrt(var+eps), t = be - mu*s, and emit
//   Wf  = swizzled bf16 of diag(s) @ W      (K x N, K padded to 32)
//   b2  = t @ W + bias                      (N)
// so the next GEMM needs no per-element normalization.
// ---------------------------------------------------------------------------
__global__ void fold_bn_swz(const float* __restrict__ stats, float invR,
                            const float* __restrict__ g, const float* __restrict__ be,
                            const float* __restrict__ W, const float* __restrict__ bias,
                            int K, int N, int KC,
                            bf16* __restrict__ Wf, float* __restrict__ bias2) {
    int idx = blockIdx.x * blockDim.x + threadIdx.x;
    int total = KC * (N / 16) * 512;
    if (idx < total) {
        int e    = idx & 15;
        int lane = (idx >> 4) & 31;
        int t    = idx >> 9;
        int NT   = N / 16;
        int nt   = t % NT;
        int kc   = t / NT;
        int k    = kc * 32 + (lane >> 4) * 8 + (e & 7) + (e >> 3) * 16;
        int n    = nt * 16 + (lane & 15);
        float v  = 0.f;
        if (k < K) {
            float mu  = stats[k] * invR;
            float var = stats[K + k] * invR - mu * mu;
            float s   = g[k] * rsqrtf(var + EPSV);
            v = W[(size_t)k * N + n] * s;
        }
        Wf[idx] = (bf16)v;
    }
    if (idx < N) {
        float acc = bias[idx];
        for (int k = 0; k < K; ++k) {
            float mu  = stats[k] * invR;
            float var = stats[K + k] * invR - mu * mu;
            float s   = g[k] * rsqrtf(var + EPSV);
            float tt  = be[k] - mu * s;
            acc += tt * W[(size_t)k * N + idx];
        }
        bias2[idx] = acc;
    }
}

// ---------------------------------------------------------------------------
// Gather/concat up to 3 sources (fp32 or bf16, optionally index-gathered)
// into a row-major bf16 A matrix with K padded (zero fill).
// Used only for the encoders (7/6-wide fp32 inputs) and node2 (fp32 agg).
// ---------------------------------------------------------------------------
struct GSrc { const void* p; int w; int isbf; const int* idx; };

__global__ __launch_bounds__(256)
void gather3(GSrc s0, GSrc s1, GSrc s2, bf16* __restrict__ A, int R, int Kp) {
    int rl  = threadIdx.x >> 5;
    int c0  = threadIdx.x & 31;
    int row = blockIdx.x * 8 + rl;
    if (row >= R) return;
    long r0 = s0.idx ? (long)s0.idx[row] : (long)row;
    long r1 = s1.p ? (s1.idx ? (long)s1.idx[row] : (long)row) : 0;
    long r2 = s2.p ? (s2.idx ? (long)s2.idx[row] : (long)row) : 0;
    for (int c = c0; c < Kp; c += 32) {
        float v = 0.f;
        if (c < s0.w) {
            v = s0.isbf ? (float)((const bf16*)s0.p)[r0 * s0.w + c]
                        : ((const float*)s0.p)[r0 * s0.w + c];
        } else if (s1.p && c < s0.w + s1.w) {
            int cc = c - s0.w;
            v = s1.isbf ? (float)((const bf16*)s1.p)[r1 * s1.w + cc]
                        : ((const float*)s1.p)[r1 * s1.w + cc];
        } else if (s2.p && c < s0.w + s1.w + s2.w) {
            int cc = c - s0.w - s1.w;
            v = s2.isbf ? (float)((const bf16*)s2.p)[r2 * s2.w + cc]
                        : ((const float*)s2.p)[r2 * s2.w + cc];
        }
        A[(size_t)row * Kp + c] = (bf16)v;
    }
}

// ---------------------------------------------------------------------------
// Shared GEMM epilogue: bias + optional leaky, row-masked stores, per-column
// stats (LDS then one global atomic per column), fused segment_sum atomics.
// ---------------------------------------------------------------------------
template <int N, int NS, int NT>
__device__ __forceinline__
void gemm_epilogue(v8f (&acc)[NS][NT], int rowbase, int half, int l16, int R,
                   const float* bias, int leaky,
                   bf16* outb, float* outf, float* stats, float* s_stats,
                   const int* segidx, float* segout, int tid) {
#pragma unroll
    for (int s = 0; s < NS; ++s) {
        const int sbase = rowbase + s * 16;
        int seg[8];
        if (segidx) {
#pragma unroll
            for (int i = 0; i < 8; ++i) {
                int r = sbase + half * 8 + i;
                seg[i] = (r < R) ? segidx[r] : -1;
            }
        }
#pragma unroll
        for (int nt = 0; nt < NT; ++nt) {
            const int n = nt * 16 + l16;
            const float bn = bias ? bias[n] : 0.f;
            float ssum = 0.f, ssq = 0.f;
#pragma unroll
            for (int i = 0; i < 8; ++i) {
                const int r = sbase + half * 8 + i;
                float v = acc[s][nt][i] + bn;
                if (leaky) v = (v >= 0.f) ? v : LEAK * v;
                if (r < R) {
                    ssum += v; ssq += v * v;
                    if (outb) outb[(size_t)r * N + n] = (bf16)v;
                    if (outf) outf[(size_t)r * N + n] = v;
                    if (segidx && seg[i] >= 0)
                        atomic_add_f(&segout[(size_t)seg[i] * N + n], v);
                }
            }
            if (stats) {
                atomicAdd(&s_stats[n],     ssum);
                atomicAdd(&s_stats[N + n], ssq);
            }
        }
    }
    if (stats) {
        __syncthreads();
        for (int i = tid; i < 2 * N; i += 256) atomic_add_f(&stats[i], s_stats[i]);
    }
}

// ---------------------------------------------------------------------------
// Generic WMMA GEMM over a staged A matrix:  out = act( A @ W + bias )
//   A: R x K bf16 row-major (K mult of 32); Wf swizzled, staged in LDS.
// N=64: 2 row-strips/wave (256 rows/block); N=128: 1 strip (128 rows/block).
// ---------------------------------------------------------------------------
template <int N>
__global__ __launch_bounds__(256)
void gemm_bf16_k(const bf16* __restrict__ A, int R, int K,
                 const bf16* __restrict__ Wf, const float* __restrict__ bias,
                 int leaky,
                 bf16* __restrict__ outb, float* __restrict__ outf,
                 float* __restrict__ stats,
                 const int* __restrict__ segidx, float* __restrict__ segout) {
    constexpr int NT   = N / 16;
    constexpr int NS   = (N == 64) ? 2 : 1;
    constexpr int WMAX = (N == 64) ? 12288 : 16384;
    __shared__ bf16  s_W[WMAX];
    __shared__ float s_stats[2 * N];
    const int tid = threadIdx.x;
    const int KC  = K >> 5;
    const int wtot = KC * NT * 512;
    for (int i = tid * 8; i < wtot; i += 256 * 8)
        *(v8bf*)(s_W + i) = *(const v8bf*)(Wf + i);
    if (stats)
        for (int i = tid; i < 2 * N; i += 256) s_stats[i] = 0.f;
    __syncthreads();

    const int wave = tid >> 5, lane = tid & 31;
    const int half = lane >> 4, l16 = lane & 15;
    const int rowbase = blockIdx.x * (128 * NS) + wave * (16 * NS);

    const bf16* abase[NS];
#pragma unroll
    for (int s = 0; s < NS; ++s) {
        int r = rowbase + s * 16 + l16;
        if (r >= R) r = R - 1;                // clamp; results row-masked later
        abase[s] = A + (size_t)r * K + half * 8;
    }

    v8f acc[NS][NT];
#pragma unroll
    for (int s = 0; s < NS; ++s)
#pragma unroll
        for (int nt = 0; nt < NT; ++nt)
#pragma unroll
            for (int i = 0; i < 8; ++i) acc[s][nt][i] = 0.f;

    v8bf lo[NS], hi[NS];
#pragma unroll
    for (int s = 0; s < NS; ++s) {
        lo[s] = *(const v8bf*)(abase[s]);
        hi[s] = *(const v8bf*)(abase[s] + 16);
    }
#pragma unroll 2
    for (int kc = 0; kc < KC; ++kc) {
        const int kn = (kc + 1 < KC) ? kc + 1 : kc;
        v8bf nlo[NS], nhi[NS];
#pragma unroll
        for (int s = 0; s < NS; ++s) {
            nlo[s] = *(const v8bf*)(abase[s] + kn * 32);
            nhi[s] = *(const v8bf*)(abase[s] + kn * 32 + 16);
        }
        v16bf a[NS];
#pragma unroll
        for (int s = 0; s < NS; ++s)
#pragma unroll
            for (int j = 0; j < 8; ++j) { a[s][j] = lo[s][j]; a[s][8 + j] = hi[s][j]; }

        const bf16* wp = s_W + (size_t)kc * NT * 512 + lane * 16;
        v16bf bcur = *(const v16bf*)(wp);
#pragma unroll
        for (int nt = 0; nt < NT; ++nt) {
            v16bf bnext;
            if (nt + 1 < NT) bnext = *(const v16bf*)(wp + (nt + 1) * 512);
#pragma unroll
            for (int s = 0; s < NS; ++s)
                acc[s][nt] = __builtin_amdgcn_wmma_f32_16x16x32_bf16(
                    false, a[s], false, bcur, (short)0, acc[s][nt], false, false);
            if (nt + 1 < NT) bcur = bnext;
        }
#pragma unroll
        for (int s = 0; s < NS; ++s) { lo[s] = nlo[s]; hi[s] = nhi[s]; }
    }
    gemm_epilogue<N, NS, NT>(acc, rowbase, half, l16, R, bias, leaky,
                             outb, outf, stats, s_stats, segidx, segout, tid);
}

// ---------------------------------------------------------------------------
// Fused-gather WMMA GEMM for the edge-wise layers:
//   A row r = concat of NSRC 64-wide bf16 sources, each optionally gathered
//   through an index array.  Every 32-wide K-chunk lies inside one 64-aligned
//   source block, so per chunk: src = kc>>1, column offset = (kc&1)*32.
//   K compile-time -> kc loop fully unrolled, all loads straight-line.
// No staging buffer: A is read directly from the feature tensors.
// ---------------------------------------------------------------------------
template <int N, int K, int NSRC>
__global__ __launch_bounds__(256)
void gemm_fused(const bf16* __restrict__ s0, const int* __restrict__ i0,
                const bf16* __restrict__ s1, const int* __restrict__ i1,
                const bf16* __restrict__ s2, const int* __restrict__ i2,
                int R,
                const bf16* __restrict__ Wf, const float* __restrict__ bias,
                int leaky,
                bf16* __restrict__ outb, float* __restrict__ outf,
                float* __restrict__ stats,
                const int* __restrict__ segidx, float* __restrict__ segout) {
    constexpr int NT = N / 16;
    constexpr int NS = (N == 64) ? 2 : 1;
    constexpr int KC = K / 32;
    __shared__ bf16  s_W[KC * NT * 512];
    __shared__ float s_stats[2 * N];
    const int tid = threadIdx.x;
    for (int i = tid * 8; i < KC * NT * 512; i += 256 * 8)
        *(v8bf*)(s_W + i) = *(const v8bf*)(Wf + i);
    if (stats)
        for (int i = tid; i < 2 * N; i += 256) s_stats[i] = 0.f;
    __syncthreads();

    const int wave = tid >> 5, lane = tid & 31;
    const int half = lane >> 4, l16 = lane & 15;
    const int rowbase = blockIdx.x * (128 * NS) + wave * (16 * NS);

    const bf16* base[NS][3];
#pragma unroll
    for (int s = 0; s < NS; ++s) {
        int r = rowbase + s * 16 + l16;
        if (r >= R) r = R - 1;                // clamp; results row-masked later
        base[s][0] = s0 + (size_t)(i0 ? i0[r] : r) * 64 + half * 8;
        if (NSRC > 1) base[s][1] = s1 + (size_t)(i1 ? i1[r] : r) * 64 + half * 8;
        if (NSRC > 2) base[s][2] = s2 + (size_t)(i2 ? i2[r] : r) * 64 + half * 8;
    }

    v8f acc[NS][NT];
#pragma unroll
    for (int s = 0; s < NS; ++s)
#pragma unroll
        for (int nt = 0; nt < NT; ++nt)
#pragma unroll
            for (int i = 0; i < 8; ++i) acc[s][nt][i] = 0.f;

#pragma unroll
    for (int kc = 0; kc < KC; ++kc) {
        const int src  = kc >> 1;             // compile-time after unroll
        const int coff = (kc & 1) * 32;
        v16bf a[NS];
#pragma unroll
        for (int s = 0; s < NS; ++s) {
            v8bf lo = *(const v8bf*)(base[s][src] + coff);
            v8bf hi = *(const v8bf*)(base[s][src] + coff + 16);
#pragma unroll
            for (int j = 0; j < 8; ++j) { a[s][j] = lo[j]; a[s][8 + j] = hi[j]; }
        }
        const bf16* wp = s_W + kc * NT * 512 + lane * 16;
        v16bf bcur = *(const v16bf*)(wp);
#pragma unroll
        for (int nt = 0; nt < NT; ++nt) {
            v16bf bnext;
            if (nt + 1 < NT) bnext = *(const v16bf*)(wp + (nt + 1) * 512);
#pragma unroll
            for (int s = 0; s < NS; ++s)
                acc[s][nt] = __builtin_amdgcn_wmma_f32_16x16x32_bf16(
                    false, a[s], false, bcur, (short)0, acc[s][nt], false, false);
            if (nt + 1 < NT) bcur = bnext;
        }
    }
    gemm_epilogue<N, NS, NT>(acc, rowbase, half, l16, R, bias, leaky,
                             outb, outf, stats, s_stats, segidx, segout, tid);
}

// segment_sum over batch (30 segments) with LDS accumulation.
__global__ __launch_bounds__(256)
void batch_sum(const void* __restrict__ src, int isbf, int Rn, int W,
               const int* __restrict__ batch, float* __restrict__ out,
               int outStride, int colOff) {
    extern __shared__ float acc[];  // NBATCH * W
    for (int i = threadIdx.x; i < NBATCH * W; i += 256) acc[i] = 0.f;
    __syncthreads();
    int base = blockIdx.x * 128;
    for (int i = threadIdx.x; i < 128 * W; i += 256) {
        int r = base + i / W;
        int c = i % W;
        if (r < Rn) {
            float v = isbf ? (float)((const bf16*)src)[(size_t)r * W + c]
                           : ((const float*)src)[(size_t)r * W + c];
            atomicAdd(&acc[batch[r] * W + c], v);
        }
    }
    __syncthreads();
    for (int i = threadIdx.x; i < NBATCH * W; i += 256)
        atomic_add_f(&out[(i / W) * outStride + colOff + (i % W)], acc[i]);
}

// ---------------------------------------------------------------------------
// Tiny 30-row MLP with batch-norm, one block, fp32 VALU (glob2 + heads).
// ---------------------------------------------------------------------------
struct SMLP {
    const float* W[4];
    const float* b[4];
    const float* g[3];
    const float* be[3];
    int sz[5];
    int nl;
};

__global__ __launch_bounds__(256)
void small_mlp(SMLP P, const float* __restrict__ in, float* __restrict__ out) {
    __shared__ float xb[NBATCH * 192];
    __shared__ float yb[NBATCH * 128];
    int tid = threadIdx.x;
    int w0 = P.sz[0];
    for (int i = tid; i < NBATCH * w0; i += 256) xb[i] = in[i];
    __syncthreads();
    for (int L = 0; L < P.nl; ++L) {
        int iw = P.sz[L], ow = P.sz[L + 1];
        const float* W  = P.W[L];
        const float* bs = P.b[L];
        for (int idx = tid; idx < NBATCH * ow; idx += 256) {
            int r = idx / ow, o = idx % ow;
            float a = bs[o];
            for (int k = 0; k < iw; ++k) a += xb[r * iw + k] * W[k * ow + o];
            yb[r * ow + o] = a;
        }
        __syncthreads();
        if (L < P.nl - 1) {
            for (int o = tid; o < ow; o += 256) {
                float tmp[NBATCH];
                float sum = 0.f, sq = 0.f;
                for (int r = 0; r < NBATCH; ++r) {
                    float v = yb[r * ow + o];
                    v = (v >= 0.f) ? v : LEAK * v;
                    tmp[r] = v; sum += v; sq += v * v;
                }
                float mu  = sum * (1.f / NBATCH);
                float var = sq * (1.f / NBATCH) - mu * mu;
                float s   = P.g[L][o] * rsqrtf(var + EPSV);
                float t   = P.be[L][o] - mu * s;
                for (int r = 0; r < NBATCH; ++r) xb[r * ow + o] = tmp[r] * s + t;
            }
        } else {
            for (int i = tid; i < NBATCH * ow; i += 256) out[i] = yb[i];
        }
        __syncthreads();
    }
}

// out[r][0] = num[r]/den[r] (den==null -> num[r]); out[r][1..128] = u[r][..]
__global__ void build_head_in(const float* __restrict__ num, const float* __restrict__ den,
                              const float* __restrict__ u, float* __restrict__ out) {
    int idx = blockIdx.x * blockDim.x + threadIdx.x;
    if (idx >= NBATCH * 129) return;
    int r = idx / 129, c = idx % 129;
    out[idx] = (c == 0) ? (den ? num[r] / den[r] : num[r]) : u[r * 128 + (c - 1)];
}

__global__ void concat3_k(const float* __restrict__ a, const float* __restrict__ b,
                          const float* __restrict__ c, float* __restrict__ out) {
    int idx = blockIdx.x * blockDim.x + threadIdx.x;
    if (idx >= NBATCH * 192) return;
    int r = idx / 192, cc = idx % 192;
    out[idx] = (cc < 64) ? a[r * 64 + cc]
             : (cc < 128) ? b[r * 64 + cc - 64]
                          : c[r * 64 + cc - 128];
}

// ---------------------------------------------------------------------------
// Host orchestration
// ---------------------------------------------------------------------------
extern "C" void kernel_launch(void* const* d_in, const int* in_sizes, int n_in,
                              void* d_out, int out_size, void* d_ws, size_t ws_size,
                              hipStream_t stream) {
    (void)in_sizes; (void)n_in; (void)out_size; (void)ws_size;
    const int NN = 60000, NE = 500000;

    auto F = [&](int i) { return (const float*)d_in[i]; };
    auto I = [&](int i) { return (const int*)d_in[i]; };

    // ---- workspace carve -------------------------------------------------
    char* ws = (char*)d_ws;
    size_t off = 0;
    auto carve = [&](size_t bytes) -> char* {
        char* p = ws + off;
        off += (bytes + 255) & ~(size_t)255;
        return p;
    };
    bf16*  Abuf  = (bf16*)carve((size_t)NE * 32 * 2);   // encoders + node2 staging
    bf16*  Hbuf  = (bf16*)carve((size_t)NE * 128 * 2);
    bf16*  EA    = (bf16*)carve((size_t)NE * 64 * 2);
    bf16*  X     = (bf16*)carve((size_t)NN * 64 * 2);
    float* agg   = (float*)carve((size_t)NN * 64 * 4);
    float* gn    = (float*)carve((size_t)NN * 128 * 4);
    float* gu    = (float*)carve((size_t)NBATCH * 192 * 4);
    float* uu    = (float*)carve((size_t)3 * NBATCH * 128 * 4);
    float* stats = (float*)carve(256 * 4);
    bf16*  Wf2   = (bf16*)carve((size_t)16384 * 2);   // folded 2nd-layer weights
    float* bias2 = (float*)carve(128 * 4);
    float* hin   = (float*)carve((size_t)3 * NBATCH * 129 * 4);
    float* hout  = (float*)carve((size_t)3 * NBATCH * 64 * 4);
    float* din   = (float*)carve((size_t)NBATCH * 192 * 4);

    auto carveW = [&](int K, int N) -> bf16* {
        int KC = (K + 31) / 32;
        return (bf16*)carve((size_t)KC * 32 * N * 2);
    };
    auto swz = [&](int idx, int K, int N, bf16* dst) {
        int KC = (K + 31) / 32;
        int total = KC * (N / 16) * 512;
        swizzle_w<<<(total + 255) / 256, 256, 0, stream>>>(F(idx), K, N, KC, dst);
    };

    // Pre-swizzled first-layer weights (param leaf indices = jax pytree order).
    bf16* encNmol_W0 = carveW(7, 64);    swz(54, 7, 64, encNmol_W0);
    bf16* encNmem_W0 = carveW(7, 64);    swz(48, 7, 64, encNmem_W0);
    bf16* encE_W0    = carveW(6, 64);    swz(42, 6, 64, encE_W0);
    bf16* edge_W0    = carveW(192, 64);  swz(60, 192, 64, edge_W0);
    bf16* n1_W0      = carveW(128, 64);  swz(82, 128, 64, n1_W0);
    bf16* n2_W0      = carveW(128, 64);  swz(88, 128, 64, n2_W0);
    bf16* g1_W0      = carveW(128, 128); swz(66, 128, 128, g1_W0);

    auto gemmN = [&](int N, const bf16* A, int R, int K, const bf16* Wf,
                     const float* bias, int leaky,
                     bf16* ob, float* of, float* st, const int* seg, float* segout) {
        if (N == 64) {
            dim3 g((R + 255) / 256);
            gemm_bf16_k<64><<<g, 256, 0, stream>>>(A, R, K, Wf, bias, leaky,
                                                   ob, of, st, seg, segout);
        } else {
            dim3 g((R + 127) / 128);
            gemm_bf16_k<128><<<g, 256, 0, stream>>>(A, R, K, Wf, bias, leaky,
                                                    ob, of, st, seg, segout);
        }
    };
    // fold BN of hidden layer into W2/b2, then run second GEMM over Hbuf
    auto foldrun = [&](int R, int Nh, int Nout,
                       const float* g0, const float* be0,
                       const float* W1, const float* b1,
                       bf16* ob, float* of, const int* seg, float* segout) {
        int KC1 = Nh / 32;
        int total = KC1 * (Nout / 16) * 512;
        int work = total > Nout ? total : Nout;
        fold_bn_swz<<<(work + 255) / 256, 256, 0, stream>>>(
            stats, 1.f / (float)R, g0, be0, W1, b1, Nh, Nout, KC1, Wf2, bias2);
        gemmN(Nout, Hbuf, R, Nh, Wf2, bias2, 0, ob, of, nullptr, seg, segout);
    };
    // staged-A 2-layer MLP (encoders, node2)
    auto mlp2 = [&](const bf16* A, int R, int K0, int Nh, int Nout,
                    const bf16* W0f, const float* b0, const float* g0, const float* be0,
                    const float* W1, const float* b1,
                    bf16* ob, float* of, const int* seg, float* segout) {
        hipMemsetAsync(stats, 0, 2 * Nh * 4, stream);
        gemmN(Nh, A, R, K0, W0f, b0, 1, Hbuf, nullptr, stats, nullptr, nullptr);
        foldrun(R, Nh, Nout, g0, be0, W1, b1, ob, of, seg, segout);
    };
    auto gath = [&](GSrc s0, GSrc s1, GSrc s2, bf16* A, int R, int Kp) {
        gather3<<<(R + 7) / 8, 256, 0, stream>>>(s0, s1, s2, A, R, Kp);
    };
    GSrc NIL{nullptr, 0, 0, nullptr};

    // ---- per-graph processing (sequential; features reused) --------------
    for (int gidx = 0; gidx < 3; ++gidx) {
        const float* xin; const float* ein; const int* ei; const int* bat;
        if (gidx == 0)      { xin = F(0); ein = F(1); ei = I(94); bat = I(95); }
        else if (gidx == 1) { xin = F(3); ein = F(4); ei = I(96); bat = I(97); }
        else                { xin = F(6); ein = F(7); ei = I(98); bat = I(99); }
        const int* row = ei;
        const int* col = ei + NE;

        // node encoder (mol for graphs 0,1; memb for graph 2)
        bf16* nW0 = (gidx == 2) ? encNmem_W0 : encNmol_W0;
        int nb = (gidx == 2) ? 48 : 54;  // W0,W1,b0,b1,be0,g0
        gath({xin, 7, 0, nullptr}, NIL, NIL, Abuf, NN, 32);
        mlp2(Abuf, NN, 32, 64, 64, nW0, F(nb + 2), F(nb + 5), F(nb + 4),
             F(nb + 1), F(nb + 3), X, nullptr, nullptr, nullptr);
        // edge encoder
        gath({ein, 6, 0, nullptr}, NIL, NIL, Abuf, NE, 32);
        mlp2(Abuf, NE, 32, 64, 64, encE_W0, F(44), F(47), F(46),
             F(43), F(45), EA, nullptr, nullptr, nullptr);

        for (int r = 0; r < 4; ++r) {
            // edge MLP: ea' = MLP([x[row], x[col], ea])  (fused gather)
            hipMemsetAsync(stats, 0, 2 * 64 * 4, stream);
            gemm_fused<64, 192, 3><<<(NE + 255) / 256, 256, 0, stream>>>(
                X, row, X, col, EA, nullptr, NE, edge_W0, F(62), 1,
                Hbuf, nullptr, stats, nullptr, nullptr);
            foldrun(NE, 64, 64, F(65), F(64), F(61), F(63),
                    EA, nullptr, nullptr, nullptr);
            // node1 MLP + fused segment_sum(m, col) -> agg  (fused gather)
            hipMemsetAsync(stats, 0, 2 * 64 * 4, stream);
            gemm_fused<64, 128, 2><<<(NE + 255) / 256, 256, 0, stream>>>(
                X, row, EA, nullptr, nullptr, nullptr, NE, n1_W0, F(84), 1,
                Hbuf, nullptr, stats, nullptr, nullptr);
            hipMemsetAsync(agg, 0, (size_t)NN * 64 * 4, stream);
            foldrun(NE, 64, 64, F(87), F(86), F(83), F(85),
                    nullptr, nullptr, col, agg);
            // node2 MLP: x' = MLP([x, agg])  (staged: agg is fp32)
            gath({X, 64, 1, nullptr}, {agg, 64, 0, nullptr}, NIL, Abuf, NN, 128);
            mlp2(Abuf, NN, 128, 64, 64, n2_W0, F(90), F(93), F(92),
                 F(89), F(91), X, nullptr, nullptr, nullptr);

            if (r == 3) {  // u only matters in the final round
                // glob1 MLP + fused segment_sum(g, col) -> gn  (fused gather)
                hipMemsetAsync(stats, 0, 2 * 128 * 4, stream);
                gemm_fused<128, 128, 2><<<(NE + 127) / 128, 256, 0, stream>>>(
                    X, row, EA, nullptr, nullptr, nullptr, NE, g1_W0, F(68), 1,
                    Hbuf, nullptr, stats, nullptr, nullptr);
                hipMemsetAsync(gn, 0, (size_t)NN * 128 * 4, stream);
                foldrun(NE, 128, 128, F(71), F(70), F(67), F(69),
                        nullptr, nullptr, col, gn);
                // gu = [segsum(x,batch) | segsum(gn,batch)]  (30 x 192)
                hipMemsetAsync(gu, 0, (size_t)NBATCH * 192 * 4, stream);
                batch_sum<<<(NN + 127) / 128, 256, NBATCH * 64 * 4, stream>>>(
                    X, 1, NN, 64, bat, gu, 192, 0);
                batch_sum<<<(NN + 127) / 128, 256, NBATCH * 128 * 4, stream>>>(
                    gn, 0, NN, 128, bat, gu, 192, 64);
                // glob2: 192 -> 128 -> 128 -> 128 with BN after layers 0,1
                SMLP P{};
                P.W[0] = F(72); P.W[1] = F(73); P.W[2] = F(74);
                P.b[0] = F(75); P.b[1] = F(76); P.b[2] = F(77);
                P.be[0] = F(78); P.be[1] = F(79);
                P.g[0] = F(80); P.g[1] = F(81);
                P.sz[0] = 192; P.sz[1] = 128; P.sz[2] = 128; P.sz[3] = 128;
                P.nl = 3;
                small_mlp<<<1, 256, 0, stream>>>(P, gu, uu + gidx * NBATCH * 128);
            }
        }
    }

    // ---- heads -----------------------------------------------------------
    int thr = (NBATCH * 129 + 255) / 256;
    build_head_in<<<thr, 256, 0, stream>>>(F(2), F(8), uu, hin);                       // [y1/ym, u1]
    build_head_in<<<thr, 256, 0, stream>>>(F(5), F(8), uu + NBATCH * 128,
                                           hin + NBATCH * 129);                        // [y2/ym, u2]
    build_head_in<<<thr, 256, 0, stream>>>(F(9), nullptr, uu + 2 * NBATCH * 128,
                                           hin + 2 * NBATCH * 129);                    // [T, um]

    auto head = [&](int base, const float* in, float* out) {  // [129,128,64], BN after L0
        SMLP P{};
        P.W[0] = F(base); P.W[1] = F(base + 1);
        P.b[0] = F(base + 2); P.b[1] = F(base + 3);
        P.be[0] = F(base + 4); P.g[0] = F(base + 5);
        P.sz[0] = 129; P.sz[1] = 128; P.sz[2] = 64;
        P.nl = 2;
        small_mlp<<<1, 256, 0, stream>>>(P, in, out);
    };
    head(10, hin, hout);                                            // a
    head(16, hin + NBATCH * 129, hout + NBATCH * 64);               // b
    head(22, hin + 2 * NBATCH * 129, hout + 2 * NBATCH * 64);       // c

    concat3_k<<<(NBATCH * 192 + 255) / 256, 256, 0, stream>>>(
        hout, hout + NBATCH * 64, hout + 2 * NBATCH * 64, din);

    SMLP Pd{};  // d: [192,128,128,128,15], BN after layers 0,1,2
    Pd.W[0] = F(28); Pd.W[1] = F(29); Pd.W[2] = F(30); Pd.W[3] = F(31);
    Pd.b[0] = F(32); Pd.b[1] = F(33); Pd.b[2] = F(34); Pd.b[3] = F(35);
    Pd.be[0] = F(36); Pd.be[1] = F(37); Pd.be[2] = F(38);
    Pd.g[0] = F(39); Pd.g[1] = F(40); Pd.g[2] = F(41);
    Pd.sz[0] = 192; Pd.sz[1] = 128; Pd.sz[2] = 128; Pd.sz[3] = 128; Pd.sz[4] = 15;
    Pd.nl = 4;
    small_mlp<<<1, 256, 0, stream>>>(Pd, din, (float*)d_out);
}
